// GLGRUCell_59261958750753
// MI455X (gfx1250) — compile-verified, matrix operand order checked by reading, not run on hardware
//
#include <hip/hip_runtime.h>
#include <stdint.h>

// ---------------------------------------------------------------------------
// GLGRU cell for MI455X (gfx1250): fp16 WMMA GEMMs with async global->LDS
// double-buffered staging + fused gate epilogues.
// ---------------------------------------------------------------------------

typedef __attribute__((ext_vector_type(16))) _Float16 v16h;
typedef __attribute__((ext_vector_type(8)))  _Float16 v8h;
typedef __attribute__((ext_vector_type(4)))  _Float16 v4h;
typedef __attribute__((ext_vector_type(8)))  float    v8f;
typedef __attribute__((ext_vector_type(4)))  float    v4f;

#define BATCH 8192
#define KDIM  2048      // D_IN == H == 2048, K of every GEMM
#define HDIM  2048

#define BK    64        // K-tile per staging round (2 WMMA sub-steps)
#define LDT   72        // LDS tile row stride in halves (64 data + 8 pad), 144B

union Frag16 { v16h v; v8h h[2]; };

// ---------------------------------------------------------------------------
// fp32 -> fp16 elementwise cast (4 elems / thread)
// ---------------------------------------------------------------------------
__global__ __launch_bounds__(256)
void cast_f32_to_f16(const float* __restrict__ src, _Float16* __restrict__ dst, int n4) {
    int i = blockIdx.x * 256 + threadIdx.x;
    if (i >= n4) return;
    v4f a = ((const v4f*)src)[i];
    ((v4h*)dst)[i] = __builtin_convertvector(a, v4h);
}

// ---------------------------------------------------------------------------
// W [K,N] fp32 row-major  ->  Wt [N,K] fp16 row-major  (32x32 LDS tiles)
// ---------------------------------------------------------------------------
__global__ __launch_bounds__(256)
void transpose_cast(const float* __restrict__ W, _Float16* __restrict__ Wt, int K, int N) {
    __shared__ _Float16 t[32][33];
    const int n0 = blockIdx.x * 32;
    const int k0 = blockIdx.y * 32;
    const int tx = threadIdx.x & 31;
    const int ty = threadIdx.x >> 5;           // 0..7
#pragma unroll
    for (int p = 0; p < 4; ++p) {
        int r = ty + p * 8;
        t[r][tx] = (_Float16)W[(size_t)(k0 + r) * N + (n0 + tx)];
    }
    __syncthreads();
#pragma unroll
    for (int p = 0; p < 4; ++p) {
        int r = ty + p * 8;
        Wt[(size_t)(n0 + r) * K + (k0 + tx)] = t[tx][r];
    }
}

// ---------------------------------------------------------------------------
// Async global->LDS staging of a 128 x BK fp16 tile (ASYNCcnt-tracked DMA,
// no VGPR data). Thread t moves 32 halves (64B) of row t/2.
// INST_OFFSET is added to BOTH the global and LDS address (ISA ch.10 §4.4).
// ---------------------------------------------------------------------------
__device__ __forceinline__
void stage_tile_async(_Float16* __restrict__ lds, const _Float16* __restrict__ gsrc,
                      int ld, int tid) {
    const int r  = tid >> 1;
    const int kc = (tid & 1) << 5;                       // 0 or 32 halves
    const _Float16* gp = gsrc + (size_t)r * ld + kc;
    const uint32_t  la = (uint32_t)(uintptr_t)(lds + r * LDT + kc);  // LDS byte offset
    asm volatile(
        "global_load_async_to_lds_b128 %0, %1, off\n\t"
        "global_load_async_to_lds_b128 %0, %1, off offset:16\n\t"
        "global_load_async_to_lds_b128 %0, %1, off offset:32\n\t"
        "global_load_async_to_lds_b128 %0, %1, off offset:48"
        :: "v"(la), "v"((const void*)gp) : "memory");
}

__device__ __forceinline__ void wait_async_zero() {
    asm volatile("s_wait_asynccnt 0x0" ::: "memory");
}

// ---------------------------------------------------------------------------
// Fragment load from LDS in ISA 7.12.2 16-bit A/B layout:
// lanes 0-15 : row = rbase+lane,    K in {ks+0..7, ks+16..23}
// lanes 16-31: row = rbase+lane-16, K in {ks+8..15, ks+24..31}
// ---------------------------------------------------------------------------
__device__ __forceinline__
v16h load_frag(const _Float16* __restrict__ lds, int rbase, int ks, int lane) {
    const int r  = rbase + (lane & 15);
    const int hi = (lane >> 4) << 3;   // 0 or 8
    Frag16 f;
    f.h[0] = *(const v8h*)(lds + r * LDT + ks + hi);
    f.h[1] = *(const v8h*)(lds + r * LDT + ks + 16 + hi);
    return f.v;
}

__device__ __forceinline__
v8f wmma_f16(v16h a, v16h b, v8f c) {
    return __builtin_amdgcn_wmma_f32_16x16x32_f16(false, a, false, b, (short)0, c,
                                                  false, false);
}

// ---------------------------------------------------------------------------
// Tiled WMMA GEMM:  OUT[M,N] (fp32) = A1[M,K] @ Wt1[N,K]^T (+ A2 @ Wt2^T) + b1 (+ b2)
// Block tile 128x128xBK, 256 threads = 8 waves, wave tile 32(M) x 64(N).
// Double-buffered async LDS staging.
// ---------------------------------------------------------------------------
template <bool DUAL>
__global__ __launch_bounds__(256)
void gemm_wmma(const _Float16* __restrict__ A1, const _Float16* __restrict__ W1,
               const _Float16* __restrict__ A2, const _Float16* __restrict__ W2,
               const float* __restrict__ b1,  const float* __restrict__ b2,
               float* __restrict__ OUT, int N, int K) {
    __shared__ __align__(16) _Float16 As1[2][128 * LDT];
    __shared__ __align__(16) _Float16 Ws1[2][128 * LDT];
    __shared__ __align__(16) _Float16 As2[DUAL ? 2 : 1][DUAL ? 128 * LDT : 8];
    __shared__ __align__(16) _Float16 Ws2[DUAL ? 2 : 1][DUAL ? 128 * LDT : 8];

    const int tid  = threadIdx.x;
    const int lane = tid & 31;
    const int wave = tid >> 5;
    const int wm   = (wave & 3) * 32;   // wave M offset in block tile
    const int wn   = (wave >> 2) * 64;  // wave N offset in block tile
    const int bm   = blockIdx.y * 128;
    const int bn   = blockIdx.x * 128;

    v8f acc[2][4];
#pragma unroll
    for (int i = 0; i < 2; ++i)
#pragma unroll
        for (int j = 0; j < 4; ++j)
            acc[i][j] = (v8f){0.f, 0.f, 0.f, 0.f, 0.f, 0.f, 0.f, 0.f};

    const _Float16* a1 = A1 + (size_t)bm * K;
    const _Float16* w1 = W1 + (size_t)bn * K;
    const _Float16* a2 = DUAL ? A2 + (size_t)bm * K : nullptr;
    const _Float16* w2 = DUAL ? W2 + (size_t)bn * K : nullptr;

    const int nkt = K / BK;

    // Prologue: stage tile 0 into buffer 0.
    stage_tile_async(As1[0], a1, K, tid);
    stage_tile_async(Ws1[0], w1, K, tid);
    if (DUAL) {
        stage_tile_async(As2[0], a2, K, tid);
        stage_tile_async(Ws2[0], w2, K, tid);
    }

    for (int kt = 0; kt < nkt; ++kt) {
        const int cur = kt & 1;

        wait_async_zero();      // my async loads of tile kt have landed in LDS
        __syncthreads();        // everyone's landed; everyone done reading buf[1-cur]

        if (kt + 1 < nkt) {     // overlap tile kt+1 fetch with tile kt compute
            const int nb = 1 - cur;
            const size_t ko = (size_t)(kt + 1) * BK;
            stage_tile_async(As1[nb], a1 + ko, K, tid);
            stage_tile_async(Ws1[nb], w1 + ko, K, tid);
            if (DUAL) {
                stage_tile_async(As2[nb], a2 + ko, K, tid);
                stage_tile_async(Ws2[nb], w2 + ko, K, tid);
            }
        }

#pragma unroll
        for (int ks = 0; ks < BK; ks += 32) {
            v16h af[2], bf[4];
#pragma unroll
            for (int i = 0; i < 2; ++i) af[i] = load_frag(As1[cur], wm + 16 * i, ks, lane);
#pragma unroll
            for (int j = 0; j < 4; ++j) bf[j] = load_frag(Ws1[cur], wn + 16 * j, ks, lane);
#pragma unroll
            for (int i = 0; i < 2; ++i)
#pragma unroll
                for (int j = 0; j < 4; ++j)
                    acc[i][j] = wmma_f16(af[i], bf[j], acc[i][j]);

            if (DUAL) {
#pragma unroll
                for (int i = 0; i < 2; ++i) af[i] = load_frag(As2[cur], wm + 16 * i, ks, lane);
#pragma unroll
                for (int j = 0; j < 4; ++j) bf[j] = load_frag(Ws2[cur], wn + 16 * j, ks, lane);
#pragma unroll
                for (int i = 0; i < 2; ++i)
#pragma unroll
                    for (int j = 0; j < 4; ++j)
                        acc[i][j] = wmma_f16(af[i], bf[j], acc[i][j]);
            }
        }
    }

    // Epilogue. C/D layout (ISA 7.12.2): VGPR e -> lanes 0-15: M=e, N=lane;
    //                                    lanes 16-31: M=8+e, N=lane-16.
    const int ncol_l = lane & 15;
    const int mplus  = (lane >> 4) * 8;
#pragma unroll
    for (int j = 0; j < 4; ++j) {
        const int ncol = bn + wn + 16 * j + ncol_l;
        float bias = (b1 ? b1[ncol] : 0.f) + (b2 ? b2[ncol] : 0.f);
#pragma unroll
        for (int i = 0; i < 2; ++i) {
#pragma unroll
            for (int e = 0; e < 8; ++e) {
                const int row = bm + wm + 16 * i + mplus + e;
                OUT[(size_t)row * N + ncol] = acc[i][j][e] + bias;
            }
        }
    }
}

// ---------------------------------------------------------------------------
// Stage-1 gates: z,r,u -> h_new (fp32 out) and h_d (fp16 for next GEMM)
// Vectorized x4 (pure streaming).
// ---------------------------------------------------------------------------
__device__ __forceinline__ float sigmoidf(float v) {
    return 1.0f / (1.0f + __expf(-v));
}

__global__ __launch_bounds__(256)
void ew_stage1(const float* __restrict__ ox,   // [B, 4H]  x@Wx + bx
               const float* __restrict__ ohg,  // [B, 3H]  h@Wh + g@Wgh + bh + bgh
               const float* __restrict__ h, const float* __restrict__ g,
               float* __restrict__ h_out, _Float16* __restrict__ hd16) {
    const int i4  = blockIdx.x * 256 + threadIdx.x;     // < B*H/4
    const int i   = i4 * 4;
    const int row = i >> 11;                            // /2048
    const int col = i & 2047;
    const size_t b4 = (size_t)row * (4 * HDIM);
    const size_t b3 = (size_t)row * (3 * HDIM);

    const v4f xz = *(const v4f*)(ox + b4 + col);
    const v4f xr = *(const v4f*)(ox + b4 + HDIM + col);
    const v4f xh = *(const v4f*)(ox + b4 + 2 * HDIM + col);
    const v4f xu = *(const v4f*)(ox + b4 + 3 * HDIM + col);
    const v4f hz = *(const v4f*)(ohg + b3 + col);
    const v4f hr = *(const v4f*)(ohg + b3 + HDIM + col);
    const v4f hu = *(const v4f*)(ohg + b3 + 2 * HDIM + col);
    const v4f hv = *(const v4f*)(h + i);
    const v4f gv = *(const v4f*)(g + i);

    v4f hn;
    v4h hd;
#pragma unroll
    for (int e = 0; e < 4; ++e) {
        const float z = sigmoidf(xz[e] + hz[e]);
        const float r = sigmoidf(xr[e] + hr[e]);
        const float u = sigmoidf(xu[e] + hu[e]);
        const float cand = tanhf(xh[e] + r * hv[e] + u * gv[e]);
        hn[e] = (1.0f - z) * hv[e] + z * cand;
        hd[e] = (_Float16)(hn[e] - hv[e]);
    }
    *(v4f*)(h_out + i) = hn;
    *(v4h*)(hd16 + i)  = hd;
}

// ---------------------------------------------------------------------------
// Stage-2 gates: z2,r2 -> g_new (vectorized x4)
// ---------------------------------------------------------------------------
__global__ __launch_bounds__(256)
void ew_stage2(const float* __restrict__ og2,  // [B, 2H]  g@Wg + bg
               const float* __restrict__ ohd,  // [B, 3H]  h_d@Whd + bhd
               const float* __restrict__ g,
               float* __restrict__ g_out) {
    const int i4  = blockIdx.x * 256 + threadIdx.x;
    const int i   = i4 * 4;
    const int row = i >> 11;
    const int col = i & 2047;
    const size_t b2 = (size_t)row * (2 * HDIM);
    const size_t b3 = (size_t)row * (3 * HDIM);

    const v4f gz2 = *(const v4f*)(og2 + b2 + col);
    const v4f gr2 = *(const v4f*)(og2 + b2 + HDIM + col);
    const v4f hdz = *(const v4f*)(ohd + b3 + col);
    const v4f hdr = *(const v4f*)(ohd + b3 + HDIM + col);
    const v4f hdg = *(const v4f*)(ohd + b3 + 2 * HDIM + col);
    const v4f gv  = *(const v4f*)(g + i);

    v4f gn;
#pragma unroll
    for (int e = 0; e < 4; ++e) {
        const float z2 = sigmoidf(hdz[e] + gz2[e]);
        const float r2 = sigmoidf(hdr[e] + gr2[e]);
        gn[e] = (1.0f - z2) * gv[e] + z2 * tanhf(hdg[e] + r2 * gv[e]);
    }
    *(v4f*)(g_out + i) = gn;
}

// ---------------------------------------------------------------------------
// Host launcher
// ---------------------------------------------------------------------------
static inline size_t align256(size_t x) { return (x + 255) & ~(size_t)255; }

extern "C" void kernel_launch(void* const* d_in, const int* in_sizes, int n_in,
                              void* d_out, int out_size, void* d_ws, size_t ws_size,
                              hipStream_t stream) {
    (void)in_sizes; (void)n_in; (void)out_size; (void)ws_size;

    const float* x   = (const float*)d_in[0];
    const float* h   = (const float*)d_in[1];
    const float* g   = (const float*)d_in[2];
    const float* Wx  = (const float*)d_in[3];
    const float* bx  = (const float*)d_in[4];
    const float* Wh  = (const float*)d_in[5];
    const float* bh  = (const float*)d_in[6];
    const float* Wgh = (const float*)d_in[7];
    const float* bgh = (const float*)d_in[8];
    const float* Whd = (const float*)d_in[9];
    const float* bhd = (const float*)d_in[10];
    const float* Wg  = (const float*)d_in[11];
    const float* bg  = (const float*)d_in[12];

    float* h_out = (float*)d_out;
    float* g_out = h_out + (size_t)BATCH * HDIM;

    const int N4 = 4 * HDIM, N3 = 3 * HDIM, N2 = 2 * HDIM;

    // workspace carve-up
    char* ws = (char*)d_ws;
    size_t off = 0;
    auto carve = [&](size_t bytes) { void* p = ws + off; off = align256(off + bytes); return p; };

    _Float16* x16   = (_Float16*)carve((size_t)BATCH * KDIM * 2);
    _Float16* h16   = (_Float16*)carve((size_t)BATCH * KDIM * 2);
    _Float16* g16   = (_Float16*)carve((size_t)BATCH * KDIM * 2);
    _Float16* hd16  = (_Float16*)carve((size_t)BATCH * KDIM * 2);
    _Float16* WxT   = (_Float16*)carve((size_t)KDIM * N4 * 2);
    _Float16* WhT   = (_Float16*)carve((size_t)KDIM * N3 * 2);
    _Float16* WghT  = (_Float16*)carve((size_t)KDIM * N3 * 2);
    _Float16* WhdT  = (_Float16*)carve((size_t)KDIM * N3 * 2);
    _Float16* WgT   = (_Float16*)carve((size_t)KDIM * N2 * 2);
    float*    out_x  = (float*)carve((size_t)BATCH * N4 * 4);
    float*    out_hg = (float*)carve((size_t)BATCH * N3 * 4);
    float*    out_g2 = (float*)carve((size_t)BATCH * N2 * 4);
    float*    out_hd = (float*)carve((size_t)BATCH * N3 * 4);

    const int actN4 = (BATCH * KDIM) / 4;

    // 1) fp16 conversions of activations
    cast_f32_to_f16<<<actN4 / 256, 256, 0, stream>>>(x, x16, actN4);
    cast_f32_to_f16<<<actN4 / 256, 256, 0, stream>>>(h, h16, actN4);
    cast_f32_to_f16<<<actN4 / 256, 256, 0, stream>>>(g, g16, actN4);

    // 2) transposed fp16 weights [N,K] (fp16 weight set totals ~126MB -> L2-resident)
    transpose_cast<<<dim3(N4 / 32, KDIM / 32), 256, 0, stream>>>(Wx,  WxT,  KDIM, N4);
    transpose_cast<<<dim3(N3 / 32, KDIM / 32), 256, 0, stream>>>(Wh,  WhT,  KDIM, N3);
    transpose_cast<<<dim3(N3 / 32, KDIM / 32), 256, 0, stream>>>(Wgh, WghT, KDIM, N3);
    transpose_cast<<<dim3(N3 / 32, KDIM / 32), 256, 0, stream>>>(Whd, WhdT, KDIM, N3);
    transpose_cast<<<dim3(N2 / 32, KDIM / 32), 256, 0, stream>>>(Wg,  WgT,  KDIM, N2);

    // 3) stage-1 GEMMs
    gemm_wmma<false><<<dim3(N4 / 128, BATCH / 128), 256, 0, stream>>>(
        x16, WxT, nullptr, nullptr, bx, nullptr, out_x, N4, KDIM);
    gemm_wmma<true><<<dim3(N3 / 128, BATCH / 128), 256, 0, stream>>>(
        h16, WhT, g16, WghT, bh, bgh, out_hg, N3, KDIM);
    gemm_wmma<false><<<dim3(N2 / 128, BATCH / 128), 256, 0, stream>>>(
        g16, WgT, nullptr, nullptr, bg, nullptr, out_g2, N2, KDIM);

    // 4) stage-1 gates -> h_new, h_d(fp16)
    ew_stage1<<<(BATCH * HDIM) / (4 * 256), 256, 0, stream>>>(out_x, out_hg, h, g, h_out, hd16);

    // 5) stage-2 GEMM on the delta
    gemm_wmma<false><<<dim3(N3 / 128, BATCH / 128), 256, 0, stream>>>(
        hd16, WhdT, nullptr, nullptr, bhd, nullptr, out_hd, N3, KDIM);

    // 6) stage-2 gates -> g_new
    ew_stage2<<<(BATCH * HDIM) / (4 * 256), 256, 0, stream>>>(out_g2, out_hd, g, g_out);
}
